// ReciprocalRankLoss_49512382988772
// MI455X (gfx1250) — matrix operand order repaired and checked by reading, not compiled, not run on hardware
//
#include <hip/hip_runtime.h>
#include <hip/hip_bf16.h>
#include <math.h>

typedef __attribute__((ext_vector_type(16))) _Float16 v16h;
typedef __attribute__((ext_vector_type(8)))  float    v8f;

#define BATCH 256
#define NPOS  4
#define NNEG  1024
#define DIM   512
#define ROWS  (NPOS + NNEG)      // 1028
#define TILES 65                 // ceil(1028/16)
#define TEMP_INV 10.0f           // 1 / 0.1
#define POSW  0.3f
#define EPSV  1e-8f

// ---------------- Kernel 1: WMMA dot products + row sumsq ----------------
// One wave = one (batch, 16-row tile). A = 16 candidate rows x 32 K-chunk (f16),
// B = anchor chunk replicated across all 16 columns (f16), C accumulates f32.
__global__ void __launch_bounds__(256)
rr_sim_wmma_kernel(const float* __restrict__ anchor,
                   const float* __restrict__ positives,
                   const float* __restrict__ negatives,
                   float* __restrict__ dots,
                   float* __restrict__ rowsq) {
    const int lane = threadIdx.x & 31;
    const int wid  = (blockIdx.x * blockDim.x + threadIdx.x) >> 5;
    const int b    = wid / TILES;
    const int tile = wid % TILES;

    const int m  = lane & 15;             // row within tile
    const int j  = tile * 16 + m;         // global row index (may be >= ROWS)
    const int jc = (j < ROWS) ? j : (ROWS - 1);

    const float* row = (jc < NPOS)
        ? (positives + ((size_t)b * NPOS + jc) * DIM)
        : (negatives + ((size_t)b * NNEG + (jc - NPOS)) * DIM);
    const float* arow = anchor + (size_t)b * DIM;

    const int hi  = (lane >> 4) & 1;      // lane group
    const int kbA = hi * 8;               // A: K {kbA..kbA+7, kbA+16..kbA+23}
    const int kbB = hi * 8;               // B: K {kbB..kbB+7, kbB+16..kbB+23}

    v8f  c  = {};
    float sq = 0.0f;

    for (int kk = 0; kk < DIM; kk += 32) {
        if (kk + 32 < DIM) __builtin_prefetch(row + kk + 32, 0, 3);

        const float4* pr0 = (const float4*)(row + kk + kbA);        // K kbA..kbA+7
        const float4* pr1 = (const float4*)(row + kk + kbA + 16);   // K kbA+16..kbA+23
        float4 a0 = pr0[0], a1 = pr0[1];
        float4 a2 = pr1[0], a3 = pr1[1];

        const float4* pb = (const float4*)(arow + kk + kbB);        // anchor chunk
        const float4* pb2 = (const float4*)(arow + kk + kbB + 16);
        float4 b0 = pb[0], b1 = pb[1], b2 = pb2[0], b3 = pb2[1];

        // row sum-of-squares (f32, exact inputs)
        sq += a0.x*a0.x + a0.y*a0.y + a0.z*a0.z + a0.w*a0.w;
        sq += a1.x*a1.x + a1.y*a1.y + a1.z*a1.z + a1.w*a1.w;
        sq += a2.x*a2.x + a2.y*a2.y + a2.z*a2.z + a2.w*a2.w;
        sq += a3.x*a3.x + a3.y*a3.y + a3.z*a3.z + a3.w*a3.w;

        v16h a, bv;
        a[0]=(_Float16)a0.x;  a[1]=(_Float16)a0.y;  a[2]=(_Float16)a0.z;  a[3]=(_Float16)a0.w;
        a[4]=(_Float16)a1.x;  a[5]=(_Float16)a1.y;  a[6]=(_Float16)a1.z;  a[7]=(_Float16)a1.w;
        a[8]=(_Float16)a2.x;  a[9]=(_Float16)a2.y;  a[10]=(_Float16)a2.z; a[11]=(_Float16)a2.w;
        a[12]=(_Float16)a3.x; a[13]=(_Float16)a3.y; a[14]=(_Float16)a3.z; a[15]=(_Float16)a3.w;

        bv[0]=(_Float16)b0.x;  bv[1]=(_Float16)b0.y;  bv[2]=(_Float16)b0.z;  bv[3]=(_Float16)b0.w;
        bv[4]=(_Float16)b1.x;  bv[5]=(_Float16)b1.y;  bv[6]=(_Float16)b1.z;  bv[7]=(_Float16)b1.w;
        bv[8]=(_Float16)b2.x;  bv[9]=(_Float16)b2.y;  bv[10]=(_Float16)b2.z; bv[11]=(_Float16)b2.w;
        bv[12]=(_Float16)b3.x; bv[13]=(_Float16)b3.y; bv[14]=(_Float16)b3.z; bv[15]=(_Float16)b3.w;

        c = __builtin_amdgcn_wmma_f32_16x16x32_f16(false, a, false, bv,
                                                   (short)0, c, false, false);
    }

    // combine the two K half-groups: lane L and lane L^16 hold the same row
    sq += __shfl_xor(sq, 16, 32);
    if (lane < 16 && j < ROWS)
        rowsq[(size_t)b * ROWS + j] = sq;

    // column 0 of C: lane 0 -> rows 0..7 in c[0..7]; lane 16 -> rows 8..15
    if ((lane & 15) == 0) {
        const int rbase = hi * 8;
        const int lim = ROWS - tile * 16 - rbase;   // valid rows remaining
        float* dbase = dots + (size_t)b * ROWS + tile * 16 + rbase;
        #pragma unroll
        for (int r = 0; r < 8; ++r)
            if (r < lim) dbase[r] = c[r];
    }
}

// ---------------- Kernel 2: per-batch softmax / rank / expected RR ----------------
__global__ void __launch_bounds__(256)
rr_batch_kernel(const float* __restrict__ anchor,
                const float* __restrict__ dots,
                const float* __restrict__ rowsq,
                float* __restrict__ err,
                float* __restrict__ possum) {
    __shared__ float s_sim[ROWS];
    __shared__ float red[256];
    __shared__ float s_rank[NPOS];

    const int b = blockIdx.x;
    const int t = threadIdx.x;

    // anchor norm
    float asq = 0.0f;
    for (int d = t; d < DIM; d += 256) {
        float v = anchor[(size_t)b * DIM + d];
        asq += v * v;
    }
    red[t] = asq; __syncthreads();
    for (int s = 128; s > 0; s >>= 1) {
        if (t < s) red[t] += red[t + s];
        __syncthreads();
    }
    const float anorm = sqrtf(red[0]);
    __syncthreads();

    // cosine sims
    for (int jj = t; jj < ROWS; jj += 256) {
        float den = fmaxf(anorm * sqrtf(rowsq[(size_t)b * ROWS + jj]), EPSV);
        s_sim[jj] = dots[(size_t)b * ROWS + jj] / den;
    }
    __syncthreads();

    // softmax max over logits = sim * 10
    float mx = -1e30f;
    for (int jj = t; jj < ROWS; jj += 256)
        mx = fmaxf(mx, s_sim[jj] * TEMP_INV);
    red[t] = mx; __syncthreads();
    for (int s = 128; s > 0; s >>= 1) {
        if (t < s) red[t] = fmaxf(red[t], red[t + s]);
        __syncthreads();
    }
    mx = red[0]; __syncthreads();

    // softmax denominator
    float se = 0.0f;
    for (int jj = t; jj < ROWS; jj += 256)
        se += expf(s_sim[jj] * TEMP_INV - mx);
    red[t] = se; __syncthreads();
    for (int s = 128; s > 0; s >>= 1) {
        if (t < s) red[t] += red[t + s];
        __syncthreads();
    }
    const float denom = red[0]; __syncthreads();

    // rank of each positive = #(sim_j > sim_p) + #(sim_j == sim_p, j < p)
    // (matches stable argsort(-probs) then argsort(order))
    float cnt[NPOS] = {0.f, 0.f, 0.f, 0.f};
    float sp0 = s_sim[0], sp1 = s_sim[1], sp2 = s_sim[2], sp3 = s_sim[3];
    for (int jj = t; jj < ROWS; jj += 256) {
        float sj = s_sim[jj];
        cnt[0] += (sj > sp0 || (sj == sp0 && jj < 0)) ? 1.0f : 0.0f;
        cnt[1] += (sj > sp1 || (sj == sp1 && jj < 1)) ? 1.0f : 0.0f;
        cnt[2] += (sj > sp2 || (sj == sp2 && jj < 2)) ? 1.0f : 0.0f;
        cnt[3] += (sj > sp3 || (sj == sp3 && jj < 3)) ? 1.0f : 0.0f;
    }
    #pragma unroll
    for (int p = 0; p < NPOS; ++p) {
        red[t] = cnt[p]; __syncthreads();
        for (int s = 128; s > 0; s >>= 1) {
            if (t < s) red[t] += red[t + s];
            __syncthreads();
        }
        if (t == 0) s_rank[p] = red[0];
        __syncthreads();
    }

    if (t == 0) {
        float e = 0.0f, ps = 0.0f;
        #pragma unroll
        for (int p = 0; p < NPOS; ++p) {
            float sim  = s_sim[p];
            float prob = expf(sim * TEMP_INV - mx) / denom;
            e  += prob / (s_rank[p] + 1.0f);
            ps += sim;
        }
        err[b]    = e;
        possum[b] = ps;
    }
}

// ---------------- Kernel 3: final scalar reduction ----------------
__global__ void __launch_bounds__(256)
rr_final_kernel(const float* __restrict__ err,
                const float* __restrict__ possum,
                float* __restrict__ out) {
    __shared__ float r1[256];
    __shared__ float r2[256];
    const int t = threadIdx.x;
    r1[t] = err[t];
    r2[t] = possum[t];
    __syncthreads();
    for (int s = 128; s > 0; s >>= 1) {
        if (t < s) { r1[t] += r1[t + s]; r2[t] += r2[t + s]; }
        __syncthreads();
    }
    if (t == 0) {
        float loss = -(r1[0] / (float)BATCH)
                   + POSW * (1.0f - r2[0] / (float)(BATCH * NPOS));
        out[0] = loss;
    }
}

extern "C" void kernel_launch(void* const* d_in, const int* in_sizes, int n_in,
                              void* d_out, int out_size, void* d_ws, size_t ws_size,
                              hipStream_t stream) {
    const float* anchor    = (const float*)d_in[0];
    const float* positives = (const float*)d_in[1];
    const float* negatives = (const float*)d_in[2];
    float* out = (float*)d_out;

    float* dots   = (float*)d_ws;                       // BATCH*ROWS
    float* rowsq  = dots  + (size_t)BATCH * ROWS;       // BATCH*ROWS
    float* err    = rowsq + (size_t)BATCH * ROWS;       // BATCH
    float* possum = err   + BATCH;                      // BATCH

    // 256 batches * 65 tiles = 16640 waves; 8 waves per 256-thread block
    const int blocks = (BATCH * TILES) / 8;             // 2080
    rr_sim_wmma_kernel<<<blocks, 256, 0, stream>>>(anchor, positives, negatives,
                                                   dots, rowsq);
    rr_batch_kernel<<<BATCH, 256, 0, stream>>>(anchor, dots, rowsq, err, possum);
    rr_final_kernel<<<1, 256, 0, stream>>>(err, possum, out);
}
